// VectorQuantizer_32590211842701
// MI455X (gfx1250) — matrix operand order repaired and checked by reading, not compiled, not run on hardware
//
#include <hip/hip_runtime.h>

typedef __attribute__((ext_vector_type(16))) _Float16 v16h;
typedef __attribute__((ext_vector_type(8)))  _Float16 v8h;
typedef __attribute__((ext_vector_type(8)))  float    v8f;

#define K_CODES      1024
#define D_DIM        64
#define N_ROWS       32768      // 32 * 32 * 32
#define STAGE_CODES  256
#define TILES_PER_BLOCK 8

// ---------------------------------------------------------------------------
// Kernel 1: fused distance + argmin via f16 WMMA.
// One wave = one 16-row tile of x. Codebook staged through LDS in 4 chunks.
// Score used for argmin: e2 - 2*x.e  (x2 and /D are row-constant -> no effect).
// kt loop unrolled x2: two independent WMMA chains per iteration fill the
// 4-coexec WMMA->VALU hazard slots instead of v_nops.
// ---------------------------------------------------------------------------
__global__ __launch_bounds__(256) void vq_argmin_wmma(
    const float* __restrict__ x,     // [32, 64, 32, 32]  (B, D, H, W)
    const float* __restrict__ emb,   // [1024, 64]
    float* __restrict__ idx_out)     // [32768] indices as float
{
  __shared__ __attribute__((aligned(32))) _Float16 eS[STAGE_CODES * D_DIM]; // 32KB
  __shared__ float e2S[STAGE_CODES];                                        // 1KB
  __shared__ __attribute__((aligned(32))) _Float16 xS[TILES_PER_BLOCK][16 * D_DIM]; // 16KB

  const int tid  = threadIdx.x;
  const int wave = tid >> 5;
  const int lane = tid & 31;
  const int m    = lane & 15;     // A row / B,D column within tile
  const int hi   = lane >> 4;     // lane half (selects K half / D row half)

  const int tile = blockIdx.x * TILES_PER_BLOCK + wave;  // 2048 tiles total
  const int n0   = tile * 16;
  const int bb   = n0 >> 10;      // batch (tiles never cross batch: 1024 rows/batch)
  const int hw0  = n0 & 1023;
  const float* xbase = x + bb * (D_DIM * 1024) + hw0;

  // Stage this wave's 16x64 x-tile into LDS as f16.
  {
    const int row = m;
    for (int dd = 0; dd < 32; ++dd) {
      int d = dd * 2 + hi;
      float v = xbase[d * 1024 + row];
      xS[wave][row * D_DIM + d] = (_Float16)v;
    }
  }

  float best[8];
  int   bidx[8];
#pragma unroll
  for (int r = 0; r < 8; ++r) { best[r] = 3.4e38f; bidx[r] = 0; }

  for (int s = 0; s < 4; ++s) {
    __syncthreads();
    // Cooperatively stage 256 codes as f16 (coalesced element-wise copy).
    const float* esrc = emb + s * STAGE_CODES * D_DIM;
    for (int i = tid; i < STAGE_CODES * D_DIM; i += 256)
      eS[i] = (_Float16)esrc[i];
    __syncthreads();
    // Per-code squared norm from the same f16 values used in the dot.
    {
      float s2 = 0.f;
      for (int d = 0; d < D_DIM; ++d) {
        float v = (float)eS[tid * D_DIM + d];
        s2 += v * v;
      }
      e2S[tid] = s2;
    }
    __syncthreads();

    // Prefetch next stage's codebook toward L2/WGP$ while this stage computes.
    // 256 threads x one address each, 256B apart -> covers the 64KB stage.
    if (s < 3)
      __builtin_prefetch(emb + (s + 1) * STAGE_CODES * D_DIM + tid * 64, 0, 1);

    // A fragments (loop-invariant): contiguous k order, same as B below.
    // lanes 0-15: M=lane, K = {0..15}; lanes 16-31: M=lane-16, K = {16..31}
    const v8h* xrow = (const v8h*)&xS[wave][m * D_DIM];
    v8h l0 = xrow[2 * hi + 0];
    v8h h0 = xrow[2 * hi + 1];
    v8h l1 = xrow[2 * hi + 4];
    v8h h1 = xrow[2 * hi + 5];
    v16h a0 = __builtin_shufflevector(l0, h0, 0,1,2,3,4,5,6,7,8,9,10,11,12,13,14,15);
    v16h a1 = __builtin_shufflevector(l1, h1, 0,1,2,3,4,5,6,7,8,9,10,11,12,13,14,15);

#pragma unroll 2
    for (int kt = 0; kt < STAGE_CODES / 16; ++kt) {
      // B 32x16: lane holds column n = lane&15, K half by lane group.
      const int cLoc = kt * 16 + m;
      const int kbB  = hi * 16;
      const v16h b0 = *(const v16h*)&eS[cLoc * D_DIM + kbB];        // dims 0..31 half
      const v16h b1 = *(const v16h*)&eS[cLoc * D_DIM + kbB + 32];   // dims 32..63 half
      v8f acc = {};
      acc = __builtin_amdgcn_wmma_f32_16x16x32_f16(false, a0, false, b0,
                                                   (short)0, acc, false, false);
      acc = __builtin_amdgcn_wmma_f32_16x16x32_f16(false, a1, false, b1,
                                                   (short)0, acc, false, false);
      const float e2v = e2S[cLoc];
      const int   cg  = s * STAGE_CODES + cLoc;
#pragma unroll
      for (int r = 0; r < 8; ++r) {
        float score = fmaf(-2.f, acc[r], e2v);
        if (score < best[r]) { best[r] = score; bidx[r] = cg; }  // strict <: keeps lowest idx
      }
    }
  }

  // Butterfly min-reduce across the 16 lanes holding the same 8 rows.
  // xor masks 1,2,4,8 stay within each 16-lane group of wave32.
#pragma unroll
  for (int off = 8; off >= 1; off >>= 1) {
#pragma unroll
    for (int r = 0; r < 8; ++r) {
      float ob = __shfl_xor(best[r], off, 32);
      int   oi = __shfl_xor(bidx[r], off, 32);
      if (ob < best[r] || (ob == best[r] && oi < bidx[r])) { best[r] = ob; bidx[r] = oi; }
    }
  }

  // Lanes 0 and 16 hold rows 0..7 and 8..15 respectively.
  if (m == 0) {
#pragma unroll
    for (int r = 0; r < 8; ++r)
      idx_out[n0 + hi * 8 + r] = (float)bidx[r];
  }
}

// ---------------------------------------------------------------------------
// Kernel 2: gather quantized vectors (f32 exact) + per-block partial SSE.
// Thread id == memory offset in [B, D, H, W] -> fully coalesced x/quant.
// ---------------------------------------------------------------------------
__global__ __launch_bounds__(256) void vq_quant_loss(
    const float* __restrict__ x,
    const float* __restrict__ emb,
    const float* __restrict__ idx_f,
    float* __restrict__ quant,
    float* __restrict__ partial)
{
  const int gid = blockIdx.x * 256 + threadIdx.x;   // 0 .. 2097151
  const int b   = gid >> 16;
  const int d   = (gid >> 10) & 63;
  const int hw  = gid & 1023;
  const int n   = (b << 10) | hw;

  const int idx = (int)idx_f[n];
  const float q  = emb[idx * D_DIM + d];
  const float xv = x[gid];
  quant[gid] = q;

  float diff = q - xv;
  float ss   = diff * diff;
#pragma unroll
  for (int off = 16; off >= 1; off >>= 1)
    ss += __shfl_xor(ss, off, 32);

  __shared__ float wsum[8];
  if ((threadIdx.x & 31) == 0) wsum[threadIdx.x >> 5] = ss;
  __syncthreads();
  if (threadIdx.x == 0) {
    float t = 0.f;
#pragma unroll
    for (int w = 0; w < 8; ++w) t += wsum[w];
    partial[blockIdx.x] = t;
  }
}

// ---------------------------------------------------------------------------
// Kernel 3: deterministic fixed-order final reduction.
// loss = (q_loss + 0.25*e_loss) = 1.25 * SSE / (N*D)
// ---------------------------------------------------------------------------
__global__ __launch_bounds__(256) void vq_loss_final(
    const float* __restrict__ partial,   // 8192 block partials
    float* __restrict__ loss)
{
  __shared__ float sh[256];
  const int t = threadIdx.x;
  float s = 0.f;
  for (int i = 0; i < 32; ++i) s += partial[t + i * 256];
  sh[t] = s;
  __syncthreads();
  for (int stride = 128; stride >= 1; stride >>= 1) {
    if (t < stride) sh[t] += sh[t + stride];
    __syncthreads();
  }
  if (t == 0) loss[0] = sh[0] * (1.25f / 2097152.f);
}

// ---------------------------------------------------------------------------
extern "C" void kernel_launch(void* const* d_in, const int* in_sizes, int n_in,
                              void* d_out, int out_size, void* d_ws, size_t ws_size,
                              hipStream_t stream) {
  (void)in_sizes; (void)n_in; (void)out_size; (void)ws_size;
  const float* x   = (const float*)d_in[0];   // 2097152 floats
  const float* emb = (const float*)d_in[1];   // 65536 floats

  float* out   = (float*)d_out;
  float* loss  = out;                         // [1]
  float* quant = out + 1;                     // [2097152], layout [B,D,H,W]
  float* idx_f = out + 1 + 2097152;           // [32768]
  float* partial = (float*)d_ws;              // 8192 floats scratch

  vq_argmin_wmma<<<256, 256, 0, stream>>>(x, emb, idx_f);
  vq_quant_loss<<<8192, 256, 0, stream>>>(x, emb, idx_f, quant, partial);
  vq_loss_final<<<1, 256, 0, stream>>>(partial, loss);
}
